// MultiHeadAttention_85933705658435
// MI455X (gfx1250) — matrix operand order repaired and checked by reading, not compiled, not run on hardware
//
#include <hip/hip_runtime.h>
#include <math.h>

// ---------------- types for WMMA ----------------
typedef __bf16 bf16;
typedef bf16  v16bf __attribute__((ext_vector_type(16)));
typedef bf16  v8bf  __attribute__((ext_vector_type(8)));
typedef float v8f   __attribute__((ext_vector_type(8)));

#define WMMA_BF16(a, b, c) \
  __builtin_amdgcn_wmma_f32_16x16x32_bf16(false, (a), false, (b), (short)0, (c), false, false)

// Problem constants
static constexpr int S_ = 2048, B_ = 2, E_ = 1024, H_ = 16, D_ = 64;
static constexpr int ROWS = S_ * B_;        // 4096
static constexpr int NQKV = 3 * E_;         // 3072

// ---------------- DPP16 row (16-lane) reductions ----------------
// v_mov_b32_dpp row_ror:N stays within each 16-lane row, which matches the
// C/B-matrix half-wave row groups. Rotate-and-combine with 1,2,4,8 yields the
// full 16-lane reduction in every lane — pure VALU, no LDS/bpermute traffic.
template <int CTRL>
__device__ __forceinline__ float dpp_mov_f32(float x) {
  int i = __builtin_bit_cast(int, x);
  i = __builtin_amdgcn_update_dpp(0, i, CTRL, 0xf, 0xf, true);
  return __builtin_bit_cast(float, i);
}
__device__ __forceinline__ float row_max16(float x) {
  x = fmaxf(x, dpp_mov_f32<0x121>(x));  // row_ror:1
  x = fmaxf(x, dpp_mov_f32<0x122>(x));  // row_ror:2
  x = fmaxf(x, dpp_mov_f32<0x124>(x));  // row_ror:4
  x = fmaxf(x, dpp_mov_f32<0x128>(x));  // row_ror:8
  return x;
}
__device__ __forceinline__ float row_sum16(float x) {
  x += dpp_mov_f32<0x121>(x);
  x += dpp_mov_f32<0x122>(x);
  x += dpp_mov_f32<0x124>(x);
  x += dpp_mov_f32<0x128>(x);
  return x;
}

// ---------------- fragment load helpers ----------------
// A-matrix 16x32 bf16 layout: lanes 0-15 hold M=lane, K=0..15;
// lanes 16-31 hold M=lane-16, K=16..31 (contiguous, matches B). One 32-byte chunk.
__device__ __forceinline__ v16bf load_a_chunks(const bf16* p, int lane) {
  return *(const v16bf*)(p + ((lane & 16) ? 16 : 0));
}
// B-matrix 32x16 bf16 layout: lane holds N=lane&15; lanes 0-15 K=0..15,
// lanes 16-31 K=16..31, contiguous (2 elems per VGPR). One 32-byte chunk.
__device__ __forceinline__ v16bf load_b_contig(const bf16* rowbase, int lane) {
  return *(const v16bf*)(rowbase + ((lane & 16) ? 16 : 0));
}

// ---------------- LayerNorm (scale only) -> bf16 ----------------
__global__ __launch_bounds__(256) void ln_kernel(const float* __restrict__ x,
                                                 const float* __restrict__ scale,
                                                 bf16* __restrict__ out) {
  __shared__ float red[256];
  const int row = blockIdx.x;
  const int tid = threadIdx.x;
  const float4 v = ((const float4*)(x + (size_t)row * E_))[tid];
  float s = v.x + v.y + v.z + v.w;
  red[tid] = s; __syncthreads();
  for (int st = 128; st > 0; st >>= 1) { if (tid < st) red[tid] += red[tid + st]; __syncthreads(); }
  const float mu = red[0] * (1.0f / E_);
  __syncthreads();
  const float dx = v.x - mu, dy = v.y - mu, dz = v.z - mu, dw = v.w - mu;
  red[tid] = dx * dx + dy * dy + dz * dz + dw * dw; __syncthreads();
  for (int st = 128; st > 0; st >>= 1) { if (tid < st) red[tid] += red[tid + st]; __syncthreads(); }
  const float rstd = rsqrtf(red[0] * (1.0f / E_) + 1e-6f);
  const float4 sc = ((const float4*)scale)[tid];
  bf16* o = out + (size_t)row * E_ + tid * 4;
  o[0] = (bf16)(dx * rstd * sc.x);
  o[1] = (bf16)(dy * rstd * sc.y);
  o[2] = (bf16)(dz * rstd * sc.z);
  o[3] = (bf16)(dw * rstd * sc.w);
}

// ---------------- weight transpose + f32->bf16 : W[K,N] -> Wt[N,K] ----------------
__global__ __launch_bounds__(256) void transpose_conv(const float* __restrict__ W,
                                                      bf16* __restrict__ Wt,
                                                      int K, int N) {
  const int idx = blockIdx.x * 256 + threadIdx.x;   // over K*N
  const int k = idx / N;
  const int n = idx - k * N;
  Wt[(size_t)n * K + k] = (bf16)W[idx];
}

// ---------------- generic GEMM: C[M,N] = A[M,K] * Bt[N,K]^T ----------------
// Block: 256 threads (8 waves), tile 128x128. Wave tile: 64 (M) x 32 (N).
template <bool F32OUT>
__global__ __launch_bounds__(256) void gemm_bf16(const bf16* __restrict__ A,
                                                 const bf16* __restrict__ Bt,
                                                 void* __restrict__ Cout,
                                                 int M, int N, int K) {
  const int tid = threadIdx.x;
  const int l = tid & 31;
  const int w = tid >> 5;
  const int wr = w & 1, wc = w >> 1;
  const int bm = blockIdx.x * 128 + wr * 64;
  const int bn = blockIdx.y * 128 + wc * 32;

  v8f acc[4][2];
  const v8f vz = {};
#pragma unroll
  for (int mt = 0; mt < 4; ++mt)
#pragma unroll
    for (int nt = 0; nt < 2; ++nt) acc[mt][nt] = vz;

  const int lm = l & 15;
  for (int k0 = 0; k0 < K; k0 += 32) {
    v16bf afr[4], bfr[2];
#pragma unroll
    for (int mt = 0; mt < 4; ++mt) {
      const bf16* ap = A + (size_t)(bm + mt * 16 + lm) * K + k0;
      afr[mt] = load_a_chunks(ap, l);
      if (k0 + 32 < K) __builtin_prefetch(ap + 32, 0, 3);   // global_prefetch_b8
    }
#pragma unroll
    for (int nt = 0; nt < 2; ++nt) {
      const bf16* bp = Bt + (size_t)(bn + nt * 16 + lm) * K + k0;
      bfr[nt] = load_b_contig(bp, l);
      if (k0 + 32 < K) __builtin_prefetch(bp + 32, 0, 3);
    }
#pragma unroll
    for (int mt = 0; mt < 4; ++mt)
#pragma unroll
      for (int nt = 0; nt < 2; ++nt)
        acc[mt][nt] = WMMA_BF16(afr[mt], bfr[nt], acc[mt][nt]);
  }

  // C layout: lane holds N = lane&15; VGPR j holds M = j (lanes<16) / 8+j (lanes>=16)
  const int rhalf = (l & 16) ? 8 : 0;
#pragma unroll
  for (int mt = 0; mt < 4; ++mt)
#pragma unroll
    for (int nt = 0; nt < 2; ++nt) {
      const int col = bn + nt * 16 + lm;
#pragma unroll
      for (int j = 0; j < 8; ++j) {
        const int row = bm + mt * 16 + rhalf + j;
        const float val = acc[mt][nt][j];
        if (F32OUT) ((float*)Cout)[(size_t)row * N + col] = val;
        else        ((bf16*)Cout)[(size_t)row * N + col] = (bf16)val;
      }
    }
}

// ---------------- flash attention: qkv bf16 [4096, 3072] -> ctx bf16 [4096, 1024] ----------------
// grid: (S/128, B*H). Block: 256 threads; wave w owns query rows q0..q0+15.
__global__ __launch_bounds__(256) void attn_kernel(const bf16* __restrict__ qkv,
                                                   bf16* __restrict__ ctxout) {
  __shared__ bf16 kbuf[32 * 64];    // K block, row-major [key][d]
  __shared__ bf16 vtbuf[64 * 32];   // V block transposed [d][key]
  __shared__ bf16 pbuf[8 * 16 * 32]; // per-wave P tiles [16][32]

  const int tid = threadIdx.x;
  const int l = tid & 31;
  const int w = tid >> 5;
  const int bh = blockIdx.y;
  const int b = bh >> 4;            // H_=16
  const int h = bh & 15;
  const int ld = NQKV;              // 3072
  const bf16* qp = qkv + h * D_;
  const bf16* kp = qkv + E_ + h * D_;
  const bf16* vp = qkv + 2 * E_ + h * D_;

  const int q0 = blockIdx.x * 128 + w * 16;
  const int lm = l & 15;
  const int rhalf = (l & 16) ? 8 : 0;

  // persistent Q fragments (16 rows x 64 d) = two A-frags (K-steps of 32)
  const size_t qrow = (size_t)((q0 + lm) * B_ + b) * ld;
  v16bf qf0 = load_a_chunks(qp + qrow + 0, l);
  v16bf qf1 = load_a_chunks(qp + qrow + 32, l);

  float m[8], lsum[8];
  v8f ctx[4];
  const v8f vz = {};
#pragma unroll
  for (int j = 0; j < 8; ++j) { m[j] = -__builtin_inff(); lsum[j] = 0.0f; }
#pragma unroll
  for (int nt = 0; nt < 4; ++nt) ctx[nt] = vz;

  // staging indices: 256 threads cover 32 keys x 8 chunks of 8 elems
  const int skey = tid >> 3;
  const int sd = (tid & 7) * 8;

  for (int kb = 0; kb < S_; kb += 32) {
    __syncthreads();
    {
      const size_t r = (size_t)((kb + skey) * B_ + b) * ld;
      const v8bf kk = *(const v8bf*)(kp + r + sd);
      const v8bf vv = *(const v8bf*)(vp + r + sd);
      *(v8bf*)(kbuf + skey * 64 + sd) = kk;
#pragma unroll
      for (int e = 0; e < 8; ++e) vtbuf[(sd + e) * 32 + skey] = vv[e];
    }
    __syncthreads();

    // scores: S[16 x 32] = Q(16x64) * K^T(64x32), two N-tiles, K-steps of 32
    v8f scr[2];
#pragma unroll
    for (int nt = 0; nt < 2; ++nt) {
      const bf16* krow = kbuf + (nt * 16 + lm) * 64;
      v16bf b0 = load_b_contig(krow + 0, l);
      v16bf b1 = load_b_contig(krow + 32, l);
      v8f z = vz;
      z = WMMA_BF16(qf0, b0, z);
      scr[nt] = WMMA_BF16(qf1, b1, z);
    }

    // online softmax update (rows live in VGPR index j; 16 lanes per row-group)
    // All reductions are DPP16 row_ror rotate-combines: no LDS traffic.
    float p0[8], p1[8];
#pragma unroll
    for (int j = 0; j < 8; ++j) {
      const float s0 = scr[0][j], s1 = scr[1][j];
      const float t = row_max16(fmaxf(s0, s1));
      const float mn = fmaxf(m[j], t);
      const float al = __expf(m[j] - mn);
      p0[j] = __expf(s0 - mn);
      p1[j] = __expf(s1 - mn);
      const float rs = row_sum16(p0[j] + p1[j]);
      lsum[j] = lsum[j] * al + rs;
      m[j] = mn;
#pragma unroll
      for (int nt = 0; nt < 4; ++nt) ctx[nt][j] *= al;
    }

    // P (C layout) -> LDS -> reload in A layout
    bf16* pw = pbuf + w * (16 * 32);
#pragma unroll
    for (int j = 0; j < 8; ++j) {
      pw[(rhalf + j) * 32 + lm] = (bf16)p0[j];
      pw[(rhalf + j) * 32 + 16 + lm] = (bf16)p1[j];
    }
    asm volatile("s_wait_dscnt 0" ::: "memory");
    const v16bf pf = load_a_chunks(pw + lm * 32, l);

    // ctx += P(16x32) * V(32x64): 4 N-tiles
#pragma unroll
    for (int nt = 0; nt < 4; ++nt) {
      const v16bf vb = load_b_contig(vtbuf + (nt * 16 + lm) * 32, l);
      ctx[nt] = WMMA_BF16(pf, vb, ctx[nt]);
    }
  }

  // epilogue: divide by row sums, write ctx bf16 [row = s*B+b, col = h*64+d]
#pragma unroll
  for (int nt = 0; nt < 4; ++nt) {
    const int col = h * D_ + nt * 16 + lm;
#pragma unroll
    for (int j = 0; j < 8; ++j) {
      const int s = q0 + rhalf + j;
      const float val = ctx[nt][j] / lsum[j];
      ctxout[(size_t)(s * B_ + b) * E_ + col] = (bf16)val;
    }
  }
}

// ---------------- launch ----------------
extern "C" void kernel_launch(void* const* d_in, const int* in_sizes, int n_in,
                              void* d_out, int out_size, void* d_ws, size_t ws_size,
                              hipStream_t stream) {
  const float* x        = (const float*)d_in[0];  // [S,B,E]
  const float* ln_scale = (const float*)d_in[1];  // [E]
  const float* w_qkv    = (const float*)d_in[2];  // [E, 3HD]
  const float* w_out    = (const float*)d_in[3];  // [HD, E]
  float* out = (float*)d_out;                     // [S,B,E] f32

  // workspace carve-up (bf16 elements)
  bf16* ln    = (bf16*)d_ws;                        // ROWS * E      = 8 MB
  bf16* wqkvT = ln    + (size_t)ROWS * E_;          // NQKV * E      = 6 MB
  bf16* woutT = wqkvT + (size_t)NQKV * E_;          // E * E         = 2 MB
  bf16* qkv   = woutT + (size_t)E_ * E_;            // ROWS * NQKV   = 24 MB
  bf16* ctx   = qkv   + (size_t)ROWS * NQKV;        // ROWS * E      = 8 MB

  ln_kernel<<<ROWS, 256, 0, stream>>>(x, ln_scale, ln);
  transpose_conv<<<(E_ * NQKV) / 256, 256, 0, stream>>>(w_qkv, wqkvT, E_, NQKV);
  transpose_conv<<<(E_ * E_) / 256, 256, 0, stream>>>(w_out, woutT, E_, E_);

  gemm_bf16<false><<<dim3(ROWS / 128, NQKV / 128), 256, 0, stream>>>(ln, wqkvT, qkv, ROWS, NQKV, E_);
  attn_kernel<<<dim3(S_ / 128, B_ * H_), 256, 0, stream>>>(qkv, ctx);
  gemm_bf16<true><<<dim3(ROWS / 128, E_ / 128), 256, 0, stream>>>(ctx, woutT, out, ROWS, E_, E_);
}
